// Attention_37031208026285
// MI455X (gfx1250) — matrix operand order repaired and checked by reading, not compiled, not run on hardware
//
#include <hip/hip_runtime.h>
#include <stdint.h>

// ---------------- problem constants ----------------
constexpr int kB   = 4;
constexpr int kS   = 2048;
constexpr int kDim = 1024;
constexpr int kH   = 16;
constexpr int kHd  = 64;           // head dim

typedef __attribute__((ext_vector_type(16))) __bf16 v16bf;
typedef __attribute__((ext_vector_type(8)))  __bf16 v8bf;
typedef __attribute__((ext_vector_type(4)))  __bf16 v4bf;
typedef __attribute__((ext_vector_type(8)))  float  v8f;

#define WMMA_BF16(a_, b_, c_) \
  __builtin_amdgcn_wmma_f32_16x16x32_bf16(false, (a_), false, (b_), (short)0, (c_), false, false)

// ---------------- CDNA5 async copy: global -> LDS, 16 bytes per lane ----------------
// VDST = LDS byte offset (low 32 bits of the generic shared pointer),
// VADDR = 64-bit global address. Tracked by ASYNCcnt.
static __device__ __forceinline__ void stage16(const __bf16* __restrict__ g, __bf16* l) {
  uint32_t lds_off = (uint32_t)(uintptr_t)l;
  uint64_t gaddr   = (uint64_t)(uintptr_t)g;
  asm volatile("global_load_async_to_lds_b128 %0, %1, off"
               :: "v"(lds_off), "v"(gaddr) : "memory");
}
static __device__ __forceinline__ void async_wait0() {
  asm volatile("s_wait_asynccnt 0" ::: "memory");
}

// ---------------- fragment loaders (per CDNA5 ISA VGPR layouts) ----------------
// A matrix 16x32 bf16 (M x K): lane m = lane&15, half = lane>>4.
// VGPR0..3 hold K = half*8 + 0..7 ; VGPR4..7 hold K = 16 + half*8 + 0..7.
static __device__ __forceinline__ v16bf load_a_frag(const __bf16* base, int ld, int row0, int k0) {
  int lane = threadIdx.x & 31;
  int m = lane & 15, half = lane >> 4;
  const __bf16* p = base + (size_t)(row0 + m) * ld + k0 + half * 8;
  union { v16bf v; v8bf h[2]; } u;
  u.h[0] = *(const v8bf*)p;
  u.h[1] = *(const v8bf*)(p + 16);
  return u.v;
}

// B matrix 32x16 bf16 (K x N) with B[k][n] = W[n0+n][k0+k] (NT form).
// VGPR0..3 hold K = kh*8 + 0..7 ; VGPR4..7 hold K = 16 + kh*8 + 0..7; lane n = lane&15.
static __device__ __forceinline__ v16bf load_b_frag_nt(const __bf16* base, int ld, int n0, int k0) {
  int lane = threadIdx.x & 31;
  int n = lane & 15, kh = lane >> 4;
  const __bf16* p = base + (size_t)(n0 + n) * ld + k0 + kh * 8;
  union { v16bf v; v8bf h[2]; } u;
  u.h[0] = *(const v8bf*)p;
  u.h[1] = *(const v8bf*)(p + 16);
  return u.v;
}

// ---------------- conversion kernels ----------------
__global__ void cvt_f32_bf16_v4(const float* __restrict__ s, __bf16* __restrict__ d, int n4) {
  int i = blockIdx.x * blockDim.x + threadIdx.x;
  if (i >= n4) return;
  float4 v = ((const float4*)s)[i];
  v4bf o;
  o.x = (__bf16)v.x; o.y = (__bf16)v.y; o.z = (__bf16)v.z; o.w = (__bf16)v.w;
  ((v4bf*)d)[i] = o;
}

// RoPE + transpose: src fp32 (b,s,h,d) -> dst bf16 (b,h,s,d)
__global__ void rope_transpose_bf16(const float* __restrict__ src, __bf16* __restrict__ dst) {
  int i = blockIdx.x * blockDim.x + threadIdx.x;   // kB*kS*kH*32 threads
  int dp = i & 31;
  int h  = (i >> 5) & (kH - 1);
  int s  = (i >> 9) & (kS - 1);
  int b  = i >> 20;
  size_t sidx = (((size_t)(b * kS + s) * kH) + h) * kHd + 2 * dp;
  float xr = src[sidx], xi = src[sidx + 1];
  float f = (float)s * __powf(10000.0f, -(float)dp * (1.0f / 32.0f));
  float c, sn;
  __sincosf(f, &sn, &c);
  size_t didx = (((size_t)(b * kH + h) * kS) + s) * kHd + 2 * dp;
  dst[didx]     = (__bf16)(xr * c - xi * sn);
  dst[didx + 1] = (__bf16)(xr * sn + xi * c);
}

// V: src fp32 (b,s,h,d) -> dst bf16 (b,h,d,s)   (s fastest -> coalesced stores)
__global__ void v_transpose_bf16(const float* __restrict__ src, __bf16* __restrict__ dst) {
  int i = blockIdx.x * blockDim.x + threadIdx.x;   // kB*kS*kH*kHd threads
  int s = i & (kS - 1);
  int d = (i >> 11) & (kHd - 1);
  int h = (i >> 17) & (kH - 1);
  int b = i >> 21;
  float v = src[(((size_t)(b * kS + s) * kH) + h) * kHd + d];
  dst[(((size_t)(b * kH + h) * kHd) + d) * kS + s] = (__bf16)v;
}

// ---------------- bf16 NT GEMM: C[M,N] fp32 = A[M,K] * W[N,K]^T ----------------
// block = 256 threads = 8 waves; block computes 128x64 C tile (wave: 16x64).
// 64x32 weight tile double-buffered in LDS via async global->LDS copies.
__global__ void gemm_nt_bf16_f32(const __bf16* __restrict__ A, const __bf16* __restrict__ W,
                                 float* __restrict__ C, int M, int N, int K) {
  int ntn  = N >> 6;
  int bm   = blockIdx.x / ntn;
  int bn   = blockIdx.x % ntn;
  int row0 = (bm << 7) + ((threadIdx.x >> 5) << 4);
  int col0 = bn << 6;

  __shared__ __bf16 wsh[2][64 * 32];
  int t  = threadIdx.x;
  int wr = t >> 2;            // 0..63 : weight row within tile
  int wc = (t & 3) << 3;      // 0,8,16,24 : k within tile (8 bf16 = 16B)

  stage16(W + (size_t)(col0 + wr) * K + wc, &wsh[0][wr * 32 + wc]);
  async_wait0();
  __syncthreads();

  v8f acc[4] = {};
  int buf = 0;
  for (int k0 = 0; k0 < K; k0 += 32) {
    if (k0 + 32 < K) {
      stage16(W + (size_t)(col0 + wr) * K + (k0 + 32) + wc, &wsh[buf ^ 1][wr * 32 + wc]);
      int lane = threadIdx.x & 31;
      __builtin_prefetch((const void*)(A + (size_t)(row0 + (lane & 15)) * K + k0 + 32), 0, 3);
    }
    v16bf a = load_a_frag(A, K, row0, k0);
#pragma unroll
    for (int tt = 0; tt < 4; ++tt) {
      v16bf b = load_b_frag_nt(&wsh[buf][0], 32, tt << 4, 0);
      acc[tt] = WMMA_BF16(a, b, acc[tt]);
    }
    async_wait0();
    __syncthreads();
    buf ^= 1;
  }

  int lane = threadIdx.x & 31;
  int n = lane & 15, half = lane >> 4;
#pragma unroll
  for (int tt = 0; tt < 4; ++tt)
#pragma unroll
    for (int r = 0; r < 8; ++r)
      C[(size_t)(row0 + r + (half << 3)) * N + col0 + (tt << 4) + n] = acc[tt][r];
}

// ---------------- flash attention ----------------
// Q,K bf16 (b,h,s,64); Vt bf16 (b,h,64,s); Out bf16 (b,s,dim).
// block = 8 waves = 128 query rows of one (b,h); K/V tiles (32 keys) shared
// through double-buffered LDS filled with async global->LDS copies.
__global__ void attn_fa(const __bf16* __restrict__ Q, const __bf16* __restrict__ Kr,
                        const __bf16* __restrict__ Vt, const int* __restrict__ seq_lens,
                        __bf16* __restrict__ Out) {
  const int qBlocksPerBH = kS / 128;               // 16
  int bh   = blockIdx.x / qBlocksPerBH;
  int qblk = blockIdx.x % qBlocksPerBH;
  int b = bh / kH, h = bh % kH;
  int wib = threadIdx.x >> 5;
  int qt  = (qblk << 3) + wib;                     // 16-row query tile index
  int lane  = threadIdx.x & 31;
  int lhalf = lane >> 4, lcol = lane & 15;

  const __bf16* qbase = Q  + (size_t)(b * kH + h) * kS * kHd;
  const __bf16* kbase = Kr + (size_t)(b * kH + h) * kS * kHd;
  const __bf16* vbase = Vt + (size_t)(b * kH + h) * kHd * kS;
  int l = seq_lens[b];

  v16bf qa0 = load_a_frag(qbase, kHd, qt * 16, 0);
  v16bf qa1 = load_a_frag(qbase, kHd, qt * 16, 32);

  __shared__ __bf16 ksh[2][32 * 64];   // [key-in-tile][d]
  __shared__ __bf16 vsh[2][64 * 32];   // [d][key-in-tile]
  __shared__ __bf16 ptile[8][16 * 32];
  __bf16* myp = &ptile[wib][0];

  int t  = threadIdx.x;
  int kr = t >> 3, kc = (t & 7) << 3;  // K tile: 32 rows x 64 cols, 16B per thread
  int vr = t >> 2, vc = (t & 3) << 3;  // V tile: 64 rows x 32 cols, 16B per thread

  stage16(kbase + (size_t)kr * kHd + kc,  &ksh[0][kr * kHd + kc]);
  stage16(vbase + (size_t)vr * kS + vc,   &vsh[0][vr * 32 + vc]);
  async_wait0();
  __syncthreads();

  v8f acc[4] = {};
  float mrow[8], lrow[8];
#pragma unroll
  for (int r = 0; r < 8; ++r) { mrow[r] = -__builtin_inff(); lrow[r] = 0.0f; }

  const float scale = 0.125f;  // 1/sqrt(64)
  int buf = 0;

  for (int kt = 0; kt < kS; kt += 32) {
    if (kt + 32 < kS) {
      stage16(kbase + (size_t)(kt + 32 + kr) * kHd + kc, &ksh[buf ^ 1][kr * kHd + kc]);
      stage16(vbase + (size_t)vr * kS + (kt + 32) + vc,  &vsh[buf ^ 1][vr * 32 + vc]);
    }

    // scores: 16 q x 32 keys, B fragments from LDS K tile
    v8f s0 = {}, s1 = {};
    s0 = WMMA_BF16(qa0, load_b_frag_nt(&ksh[buf][0], kHd, 0, 0),  s0);
    s0 = WMMA_BF16(qa1, load_b_frag_nt(&ksh[buf][0], kHd, 0, 32), s0);
    s1 = WMMA_BF16(qa0, load_b_frag_nt(&ksh[buf][0], kHd, 16, 0),  s1);
    s1 = WMMA_BF16(qa1, load_b_frag_nt(&ksh[buf][0], kHd, 16, 32), s1);

    // scale + reference mask: -inf where (q < l) && (k >= l)
#pragma unroll
    for (int r = 0; r < 8; ++r) {
      int qpos = qt * 16 + r + (lhalf << 3);
      bool qlt = qpos < l;
      int kp0 = kt + lcol, kp1 = kt + 16 + lcol;
      s0[r] = s0[r] * scale + ((qlt && kp0 >= l) ? -__builtin_inff() : 0.0f);
      s1[r] = s1[r] * scale + ((qlt && kp1 >= l) ? -__builtin_inff() : 0.0f);
    }

    // online softmax (rows live in 16-lane halves; xor<=8 stays within a half)
#pragma unroll
    for (int r = 0; r < 8; ++r) {
      float rm = fmaxf(s0[r], s1[r]);
#pragma unroll
      for (int off = 1; off < 16; off <<= 1) rm = fmaxf(rm, __shfl_xor(rm, off, 32));
      float nm = fmaxf(mrow[r], rm);
      float alpha = __expf(mrow[r] - nm);
      mrow[r] = nm;
      float p0 = __expf(s0[r] - nm);
      float p1 = __expf(s1[r] - nm);
      s0[r] = p0; s1[r] = p1;
      float rs = p0 + p1;
#pragma unroll
      for (int off = 1; off < 16; off <<= 1) rs += __shfl_xor(rs, off, 32);
      lrow[r] = lrow[r] * alpha + rs;
      acc[0][r] *= alpha; acc[1][r] *= alpha; acc[2][r] *= alpha; acc[3][r] *= alpha;
    }

    // P (C layout) -> per-wave LDS 16x32 -> reload as A fragment
#pragma unroll
    for (int r = 0; r < 8; ++r) {
      int row = r + (lhalf << 3);
      myp[row * 32 + lcol]      = (__bf16)s0[r];
      myp[row * 32 + 16 + lcol] = (__bf16)s1[r];
    }
    asm volatile("s_wait_dscnt 0" ::: "memory");
    v16bf pa = load_a_frag(myp, 32, 0, 0);

    // out += P(16x32) * V(32x64); B fragments from LDS V tile
#pragma unroll
    for (int tt = 0; tt < 4; ++tt) {
      v16bf vb = load_b_frag_nt(&vsh[buf][0], 32, tt << 4, 0);
      acc[tt] = WMMA_BF16(pa, vb, acc[tt]);
    }

    async_wait0();
    __syncthreads();
    buf ^= 1;
  }

  // normalize and write attn output (b, s, dim) bf16
#pragma unroll
  for (int tt = 0; tt < 4; ++tt)
#pragma unroll
    for (int r = 0; r < 8; ++r) {
      int qpos = qt * 16 + r + (lhalf << 3);
      float o = acc[tt][r] / lrow[r];
      Out[(size_t)(b * kS + qpos) * kDim + h * kHd + (tt << 4) + lcol] = (__bf16)o;
    }
}

// ---------------- host side ----------------
extern "C" void kernel_launch(void* const* d_in, const int* in_sizes, int n_in,
                              void* d_out, int out_size, void* d_ws, size_t ws_size,
                              hipStream_t stream) {
  const float* x  = (const float*)d_in[0];
  const int*   sl = (const int*)d_in[1];
  const float* wq = (const float*)d_in[2];
  const float* wk = (const float*)d_in[3];
  const float* wv = (const float*)d_in[4];
  const float* wo = (const float*)d_in[5];
  float* out = (float*)d_out;

  const size_t NTOK = (size_t)kB * kS;           // 8192 tokens
  const size_t NX   = NTOK * kDim;               // 8,388,608 elements
  const size_t NW   = (size_t)kDim * kDim;       // 1,048,576 elements

  char* ws = (char*)d_ws;
  __bf16* xb  = (__bf16*)(ws);
  __bf16* wqb = (__bf16*)(ws + NX * 2);
  __bf16* wkb = wqb + NW;
  __bf16* wvb = wkb + NW;
  __bf16* wob = wvb + NW;
  float*  qf  = (float*)(ws + NX * 2 + 4 * NW * 2);
  float*  kf  = qf + NX;
  float*  vf  = kf + NX;
  __bf16* qb  = (__bf16*)(vf + NX);
  __bf16* kb  = qb + NX;
  __bf16* vt  = kb + NX;
  __bf16* attnb = (__bf16*)qf;                   // qf is dead after RoPE; reuse

  const int TPB = 256;

  // 1) fp32 -> bf16 conversions
  cvt_f32_bf16_v4<<<(int)(NX / 4 / TPB), TPB, 0, stream>>>(x,  xb,  (int)(NX / 4));
  cvt_f32_bf16_v4<<<(int)(NW / 4 / TPB), TPB, 0, stream>>>(wq, wqb, (int)(NW / 4));
  cvt_f32_bf16_v4<<<(int)(NW / 4 / TPB), TPB, 0, stream>>>(wk, wkb, (int)(NW / 4));
  cvt_f32_bf16_v4<<<(int)(NW / 4 / TPB), TPB, 0, stream>>>(wv, wvb, (int)(NW / 4));
  cvt_f32_bf16_v4<<<(int)(NW / 4 / TPB), TPB, 0, stream>>>(wo, wob, (int)(NW / 4));

  // 2) QKV projections: (8192x1024) x (1024x1024)^T, fp32 accum
  int gemm_blocks = (int)((NTOK / 128) * (kDim / 64));    // 64*16 = 1024
  gemm_nt_bf16_f32<<<gemm_blocks, TPB, 0, stream>>>(xb, wqb, qf, (int)NTOK, kDim, kDim);
  gemm_nt_bf16_f32<<<gemm_blocks, TPB, 0, stream>>>(xb, wkb, kf, (int)NTOK, kDim, kDim);
  gemm_nt_bf16_f32<<<gemm_blocks, TPB, 0, stream>>>(xb, wvb, vf, (int)NTOK, kDim, kDim);

  // 3) RoPE(Q,K) -> (b,h,s,d) bf16 ; V -> (b,h,d,s) bf16
  int rope_threads = kB * kS * kH * 32;                   // 4,194,304
  rope_transpose_bf16<<<rope_threads / TPB, TPB, 0, stream>>>(qf, qb);
  rope_transpose_bf16<<<rope_threads / TPB, TPB, 0, stream>>>(kf, kb);
  int vt_threads = kB * kS * kH * kHd;                    // 8,388,608
  v_transpose_bf16<<<vt_threads / TPB, TPB, 0, stream>>>(vf, vt);

  // 4) flash attention: 8 waves (128 query rows) per block
  int attn_blocks = kB * kH * (kS / 128);                 // 1024
  attn_fa<<<attn_blocks, TPB, 0, stream>>>(qb, kb, vt, sl, attnb);

  // 5) output projection -> fp32 d_out
  gemm_nt_bf16_f32<<<gemm_blocks, TPB, 0, stream>>>(attnb, wob, out, (int)NTOK, kDim, kDim);
}
